// GCN_90340342104697
// MI455X (gfx1250) — matrix operand order repaired and checked by reading, compiled, not run on hardware
//
#include <hip/hip_runtime.h>
#include <hip/hip_bf16.h>

#define N_NODES 2708
#define IN_C    2048
#define OUT_C   512
#define KSPLIT  4
#define KCHUNK  (IN_C / KSPLIT)                        // 512, power of two

typedef __attribute__((ext_vector_type(8)))  __bf16 v8bf;
typedef __attribute__((ext_vector_type(16))) __bf16 v16bf;
typedef __attribute__((ext_vector_type(8)))  float  v8f;

#if defined(__has_builtin) && __has_builtin(__builtin_amdgcn_sched_group_barrier)
#define SCHED_GROUP(mask, cnt, id) __builtin_amdgcn_sched_group_barrier(mask, cnt, id)
#else
#define SCHED_GROUP(mask, cnt, id)
#endif

// ---------------------------------------------------------------------------
// f32 -> bf16 streaming conversion (v_cvt_pk_bf16_f32, 16B packed stores).
// Used for x, W (pass 0) and for h -> hb (pass 1b).
// ---------------------------------------------------------------------------
__global__ void f32_to_bf16_vec8(const float* __restrict__ in,
                                 __bf16* __restrict__ out, int n8) {
    int i      = blockIdx.x * blockDim.x + threadIdx.x;
    int stride = gridDim.x * blockDim.x;
    for (; i < n8; i += stride) {
        const float4* p = (const float4*)in + 2 * (size_t)i;
        float4 a = p[0], b = p[1];
        v8bf v;
        v[0] = (__bf16)a.x; v[1] = (__bf16)a.y; v[2] = (__bf16)a.z; v[3] = (__bf16)a.w;
        v[4] = (__bf16)b.x; v[5] = (__bf16)b.y; v[6] = (__bf16)b.z; v[7] = (__bf16)b.w;
        *(v8bf*)(out + 8 * (size_t)i) = v;
    }
}

// h[m][n] = bias[n]  (K-split partials then atomically accumulate)
__global__ void gcn_h_init(const float* __restrict__ bias,
                           float* __restrict__ h, int n) {
    int i      = blockIdx.x * blockDim.x + threadIdx.x;
    int stride = gridDim.x * blockDim.x;
    for (; i < n; i += stride) h[i] = bias[i & (OUT_C - 1)];
}

__device__ __forceinline__ v16bf load_a_frag(const __bf16* __restrict__ xrow,
                                             int k0, int half) {
    v8bf lo = *(const v8bf*)(xrow + k0 + half * 16);
    v8bf hi = *(const v8bf*)(xrow + k0 + half * 16 + 8);
    return __builtin_shufflevector(lo, hi,
            0, 1, 2, 3, 4, 5, 6, 7, 8, 9, 10, 11, 12, 13, 14, 15);
}

// ---------------------------------------------------------------------------
// Pass 1: h += x @ W^T (bf16 WMMA, f32 acc). One wave owns a 16(M) x 64(N)
// tile over a 512-deep K chunk (grid.z = 4 -> 5440 waves for occupancy);
// partial results accumulated into h with global_atomic_add_f32.
// sched_group_barrier pipelines each iteration as 10 VMEM reads -> 4 WMMA,
// forcing all 4 B fragments live at once (partial waits instead of
// wait-0 before every wmma).
// ---------------------------------------------------------------------------
__global__ __launch_bounds__(128) void gcn_gemm_wmma(
    const __bf16* __restrict__ xb,   // [N_NODES][IN_C] bf16
    const __bf16* __restrict__ wb,   // [OUT_C][IN_C] bf16 (row n == B column n)
    float* __restrict__ h)           // [N_NODES][OUT_C], pre-filled with bias
{
    const int lane = threadIdx.x & 31;
    const int wave = threadIdx.x >> 5;                 // 0..3
    const int m0   = blockIdx.x * 16;
    const int n0   = (blockIdx.y * 4 + wave) * 64;     // 64 N-cols per wave
    const int kbeg = blockIdx.z * KCHUNK;

    const int half = lane >> 4;                        // 0: lanes 0-15, 1: 16-31
    const int lrow = lane & 15;

    // Clamp OOB rows to row 0: their D rows are garbage but never stored,
    // keeping the K-loop branch-free (EXEC all-1s, as WMMA requires anyway).
    const int gm = m0 + lrow;
    const int xm = (gm < N_NODES) ? gm : 0;
    const __bf16* xrow = xb + (size_t)xm * IN_C;
    const __bf16* wr0  = wb + (size_t)(n0 + lrow) * IN_C;
    const __bf16* wr1  = wr0 + (size_t)16 * IN_C;
    const __bf16* wr2  = wr0 + (size_t)32 * IN_C;
    const __bf16* wr3  = wr0 + (size_t)48 * IN_C;

    v8f acc0 = {}, acc1 = {}, acc2 = {}, acc3 = {};

    // prologue fragments for k = kbeg
    v16bf a  = load_a_frag(xrow, kbeg, half);
    int   kb = kbeg + half * 16;
    v16bf b0 = *(const v16bf*)(wr0 + kb);
    v16bf b1 = *(const v16bf*)(wr1 + kb);
    v16bf b2 = *(const v16bf*)(wr2 + kb);
    v16bf b3 = *(const v16bf*)(wr3 + kb);

    for (int k0 = kbeg; k0 < kbeg + KCHUNK; k0 += 32) {
        // next K-step loads (wrap inside chunk on last iter; value unused)
        const int kn  = kbeg + (((k0 + 32) - kbeg) & (KCHUNK - 1));
        const int kbn = kn + half * 16;
        v16bf an  = load_a_frag(xrow, kn, half);
        v16bf nb0 = *(const v16bf*)(wr0 + kbn);
        v16bf nb1 = *(const v16bf*)(wr1 + kbn);
        v16bf nb2 = *(const v16bf*)(wr2 + kbn);
        v16bf nb3 = *(const v16bf*)(wr3 + kbn);

        acc0 = __builtin_amdgcn_wmma_f32_16x16x32_bf16(false, a, false, b0,
                                                       (short)0, acc0, false, false);
        acc1 = __builtin_amdgcn_wmma_f32_16x16x32_bf16(false, a, false, b1,
                                                       (short)0, acc1, false, false);
        acc2 = __builtin_amdgcn_wmma_f32_16x16x32_bf16(false, a, false, b2,
                                                       (short)0, acc2, false, false);
        acc3 = __builtin_amdgcn_wmma_f32_16x16x32_bf16(false, a, false, b3,
                                                       (short)0, acc3, false, false);

        a = an; b0 = nb0; b1 = nb1; b2 = nb2; b3 = nb3;

        // desired per-iteration pipeline: all loads first, then the 4 WMMAs
        SCHED_GROUP(0x020, 10, 0);   // 10 x VMEM read
        SCHED_GROUP(0x008, 4, 0);    //  4 x MFMA/WMMA
    }

    // accumulate partial tile: lane holds rows {0..7}/{8..15} of one column
    const int mbase = m0 + half * 8;
    v8f accs[4] = {acc0, acc1, acc2, acc3};
#pragma unroll
    for (int t = 0; t < 4; ++t) {
        const int n = n0 + t * 16 + lrow;
#pragma unroll
        for (int r = 0; r < 8; ++r) {
            const int m = mbase + r;
            if (m < N_NODES)
                unsafeAtomicAdd(h + (size_t)m * OUT_C + n, accs[t][r]);
        }
    }
}

// ---------------------------------------------------------------------------
// Pass 2: counting sort of edges by destination -> CSR.
// ---------------------------------------------------------------------------
__global__ void zero_u32(unsigned* __restrict__ p, int n) {
    int i = blockIdx.x * blockDim.x + threadIdx.x;
    int stride = gridDim.x * blockDim.x;
    for (; i < n; i += stride) p[i] = 0u;
}

__global__ void edge_count(const int* __restrict__ dst,
                           int* __restrict__ count, int E) {
    int i = blockIdx.x * blockDim.x + threadIdx.x;
    int stride = gridDim.x * blockDim.x;
    for (; i < E; i += stride) atomicAdd(&count[dst[i]], 1);
}

// one block, 1024 threads, 3 entries/thread (3072 >= 2708)
__global__ __launch_bounds__(1024) void scan_offsets(
    const int* __restrict__ count, int* __restrict__ offsets,
    int* __restrict__ cursor)
{
    __shared__ int part[1024];
    const int t    = threadIdx.x;
    const int base = t * 3;
    int c0 = (base     < N_NODES) ? count[base]     : 0;
    int c1 = (base + 1 < N_NODES) ? count[base + 1] : 0;
    int c2 = (base + 2 < N_NODES) ? count[base + 2] : 0;
    int s  = c0 + c1 + c2;
    part[t] = s;
    __syncthreads();
    for (int off = 1; off < 1024; off <<= 1) {   // Hillis-Steele inclusive
        int v = (t >= off) ? part[t - off] : 0;
        __syncthreads();
        part[t] += v;
        __syncthreads();
    }
    const int excl = part[t] - s;
    if (base     < N_NODES) { offsets[base]     = excl;           cursor[base]     = excl; }
    if (base + 1 < N_NODES) { offsets[base + 1] = excl + c0;      cursor[base + 1] = excl + c0; }
    if (base + 2 < N_NODES) { offsets[base + 2] = excl + c0 + c1; cursor[base + 2] = excl + c0 + c1; }
    if (t == 1023) offsets[N_NODES] = part[1023];
}

__global__ void edge_place(const int* __restrict__ src, const int* __restrict__ dst,
                           int* __restrict__ cursor, int* __restrict__ sorted_src,
                           int E) {
    int i = blockIdx.x * blockDim.x + threadIdx.x;
    int stride = gridDim.x * blockDim.x;
    for (; i < E; i += stride) {
        const int d   = dst[i];
        const int pos = atomicAdd(&cursor[d], 1);
        sorted_src[pos] = src[i];
    }
}

// ---------------------------------------------------------------------------
// Pass 3: out[n] = sum_{e: dst=n} hb[src[e]].  hb is bf16 -> 1 KB gathered
// per edge instead of 2 KB (this stage carries ~0.5 GB, the pipeline's
// dominant traffic). One wave per (node, 256-col chunk); lane covers 8 cols
// via one uint4 load, unpacks bf16->f32 with shift/mask, accumulates in f32.
// ---------------------------------------------------------------------------
__device__ __forceinline__ void acc_bf16x8(float* s, uint4 u) {
    s[0] += __uint_as_float(u.x << 16);
    s[1] += __uint_as_float(u.x & 0xffff0000u);
    s[2] += __uint_as_float(u.y << 16);
    s[3] += __uint_as_float(u.y & 0xffff0000u);
    s[4] += __uint_as_float(u.z << 16);
    s[5] += __uint_as_float(u.z & 0xffff0000u);
    s[6] += __uint_as_float(u.w << 16);
    s[7] += __uint_as_float(u.w & 0xffff0000u);
}

__global__ __launch_bounds__(256) void gcn_aggregate(
    const int* __restrict__ sorted_src, const int* __restrict__ offsets,
    const __bf16* __restrict__ hb, float* __restrict__ out)
{
    const int lane = threadIdx.x & 31;
    const int w    = threadIdx.x >> 5;                 // 0..7
    const int node = blockIdx.x * 8 + w;
    if (node >= N_NODES) return;
    const int col  = blockIdx.y * 256 + lane * 8;      // 8 bf16 cols per lane

    const int beg = offsets[node];
    const int end = offsets[node + 1];

    float s[8] = {0.f, 0.f, 0.f, 0.f, 0.f, 0.f, 0.f, 0.f};
    int e = beg;
    for (; e + 2 <= end; e += 2) {                     // 2 gathers in flight
        const int n0 = sorted_src[e];
        const int n1 = sorted_src[e + 1];
        const uint4 u0 = *(const uint4*)(hb + (size_t)n0 * OUT_C + col);
        const uint4 u1 = *(const uint4*)(hb + (size_t)n1 * OUT_C + col);
        acc_bf16x8(s, u0);
        acc_bf16x8(s, u1);
    }
    if (e < end) {
        const uint4 u = *(const uint4*)(hb + (size_t)sorted_src[e] * OUT_C + col);
        acc_bf16x8(s, u);
    }

    float4 r0; r0.x = s[0]; r0.y = s[1]; r0.z = s[2]; r0.w = s[3];
    float4 r1; r1.x = s[4]; r1.y = s[5]; r1.z = s[6]; r1.w = s[7];
    float* op = out + (size_t)node * OUT_C + col;
    *(float4*)op       = r0;
    *(float4*)(op + 4) = r1;
}

extern "C" void kernel_launch(void* const* d_in, const int* in_sizes, int n_in,
                              void* d_out, int out_size, void* d_ws, size_t ws_size,
                              hipStream_t stream) {
    const float* x    = (const float*)d_in[0];
    const int*   ei   = (const int*)d_in[1];
    const float* w    = (const float*)d_in[2];
    const float* bias = (const float*)d_in[3];
    float* out = (float*)d_out;

    const int E = in_sizes[1] / 2;
    const int* src = ei;                               // edge_index[0][:]
    const int* dst = ei + E;                           // edge_index[1][:]

    // workspace layout (16B-aligned slabs), ~23.6 MB total
    char* ws = (char*)d_ws;
    size_t off = 0;
    float*  h  = (float*)(ws + off);  off += (size_t)N_NODES * OUT_C * 4;   // 5.55 MB
    __bf16* hb = (__bf16*)(ws + off); off += (size_t)N_NODES * OUT_C * 2;   // 2.77 MB
    __bf16* wb = (__bf16*)(ws + off); off += (size_t)OUT_C * IN_C * 2;      // 2 MB
    __bf16* xb = (__bf16*)(ws + off); off += (size_t)N_NODES * IN_C * 2;    // 11.1 MB
    int* sorted_src = (int*)(ws + off); off += (size_t)E * 4;               // 2 MB
    int* offsets    = (int*)(ws + off); off += ((size_t)(N_NODES + 1) * 4 + 15) & ~15ull;
    int* cursor     = (int*)(ws + off); off += ((size_t)N_NODES * 4 + 15) & ~15ull;
    int* count      = (int*)(ws + off);

    // 0) bf16 conversion of x and W; h = bias broadcast
    const int nx8 = (N_NODES * IN_C) / 8;
    const int nw8 = (OUT_C * IN_C) / 8;
    f32_to_bf16_vec8<<<(nx8 + 1023) / 1024, 256, 0, stream>>>(x, xb, nx8);
    f32_to_bf16_vec8<<<(nw8 + 1023) / 1024, 256, 0, stream>>>(w, wb, nw8);
    const int nh = N_NODES * OUT_C;
    gcn_h_init<<<(nh + 1023) / 1024, 256, 0, stream>>>(bias, h, nh);

    // 1) h += x @ W^T  (bf16 WMMA, f32 acc, K-split x4 for occupancy)
    dim3 grid((N_NODES + 15) / 16, 2, KSPLIT);         // 170 x 2 x 4, 4 waves/block
    gcn_gemm_wmma<<<grid, 128, 0, stream>>>(xb, wb, h);

    // 1b) hb = bf16(h)  -- halves the gather traffic of the dominant stage
    const int nh8 = nh / 8;
    f32_to_bf16_vec8<<<(nh8 + 1023) / 1024, 256, 0, stream>>>(h, hb, nh8);

    // 2) counting sort of edges by destination
    zero_u32<<<(N_NODES + 255) / 256, 256, 0, stream>>>((unsigned*)count, N_NODES);
    edge_count<<<1024, 256, 0, stream>>>(dst, count, E);
    scan_offsets<<<1, 1024, 0, stream>>>(count, offsets, cursor);
    edge_place<<<1024, 256, 0, stream>>>(src, dst, cursor, sorted_src, E);

    // 3) out[n] = sum_{e: dst=n} hb[src[e]]   (gather-reduce, no f32 atomics)
    dim3 agrid((N_NODES + 7) / 8, OUT_C / 256);        // 339 x 2, 8 waves/block
    gcn_aggregate<<<agrid, 256, 0, stream>>>(sorted_src, offsets, hb, out);
}
